// TemporalConvNet_28432683500176
// MI455X (gfx1250) — matrix operand order repaired and checked by reading, not compiled, not run on hardware
//
#include <hip/hip_runtime.h>

typedef __attribute__((ext_vector_type(16))) __bf16 v16bf;
typedef __attribute__((ext_vector_type(8)))  float  v8f;

__device__ __forceinline__ float eluf(float x) {
    return x > 0.0f ? x : (__expf(x) - 1.0f);
}

// CDNA5 async global->LDS copy, 16B per lane, tracked by ASYNCcnt.
//   global_load_async_to_lds_b128 vdst(lds byte addr), vaddr(64b global), off
__device__ __forceinline__ void async_copy_b128(unsigned lds_byte_off, const void* gsrc) {
    asm volatile("global_load_async_to_lds_b128 %0, %1, off"
                 :: "v"(lds_byte_off), "v"(gsrc) : "memory");
}
__device__ __forceinline__ void wait_async0() {
    asm volatile("s_wait_asynccnt 0x0" ::: "memory");
}
// Low 32 bits of a flat LDS pointer are the LDS byte address (ISA 10.2).
__device__ __forceinline__ unsigned lds_addr(const void* p) {
    return (unsigned)(size_t)p;
}

// ---------------------------------------------------------------------------
// fp32 -> bf16 elementwise convert (for the network input x)
// ---------------------------------------------------------------------------
__global__ void f32_to_bf16_kernel(const float* __restrict__ in,
                                   __bf16* __restrict__ out, int n) {
    for (int i = blockIdx.x * blockDim.x + threadIdx.x; i < n;
         i += gridDim.x * blockDim.x)
        out[i] = (__bf16)in[i];
}

// ---------------------------------------------------------------------------
// Repack conv weights W[Cout][Cin][NTAPS] (fp32) into bf16 W'[co][kvirt]
// where kvirt = tap*CIN + ci  (tap0 multiplies x[s-d], tap1 multiplies x[s]).
// ---------------------------------------------------------------------------
__global__ void prep_w_kernel(const float* __restrict__ W,
                              __bf16* __restrict__ out, int CIN, int NTAPS) {
    const int KV = CIN * NTAPS;
    const int n  = 128 * KV;
    for (int i = blockIdx.x * blockDim.x + threadIdx.x; i < n;
         i += gridDim.x * blockDim.x) {
        const int co = i / KV, k = i % KV;
        const int t = k / CIN, c = k % CIN;
        out[i] = (__bf16)W[(co * CIN + c) * NTAPS + t];
    }
}

// ---------------------------------------------------------------------------
// Fused causal dilated conv (NTAPS taps) as WMMA GEMM:
//   Yt[time, cout] = sum_k Xt[time, k] * Wt[k, cout]   (+bias/ELU/res/skip)
// virtual k < CIN -> tap0 (x[s-d]);  k >= CIN -> tap1 (x[s]).
// One WG = 256 thr = 8 waves, covers 128 time steps x 128 couts of one batch.
// Weights async-copied into LDS once per WG; activations staged time-major
// with a 16-row causal halo (max dilation 16).
// mode: 0 = bias only, 1 = bias+ELU, 2 = bias,ELU,+res,ELU, 3 = mode2 + skip
// ---------------------------------------------------------------------------
template <int CIN, int NTAPS>
__global__ __launch_bounds__(256) void tcn_conv_kernel(
    const __bf16* __restrict__ X,     // [B][CIN][S] bf16 activations
    const __bf16* __restrict__ Wt,    // [128][KV] prepacked bf16 weights
    const float*  __restrict__ bias,  // [128]
    const __bf16* __restrict__ Res,   // [B][128][S] or null
    const __bf16* __restrict__ Skip,  // [B][128][S] or null
    __bf16* __restrict__ Yb,          // bf16 out or null
    float*  __restrict__ Yf,          // fp32 out or null
    int S, int dil, int mode)
{
    constexpr int KV   = CIN * NTAPS;              // virtual reduction dim
    constexpr int NCH  = KV / 32;                  // wmma K-chunks
    constexpr int CPAD = (CIN == 128) ? 136 : 72;  // act row pitch (bf16)
    constexpr int KP   = KV + 8;                   // weight row pitch (bf16)
    __shared__ __bf16 act[144 * CPAD];             // rows: time t0-16 .. t0+127
    __shared__ __bf16 wlds[128 * KP];              // [co][kvirt] padded

    const int b    = blockIdx.y;
    const int seg  = blockIdx.x;        // 128-time segment
    const int tid  = threadIdx.x;
    const int w    = tid >> 5;          // wave id 0..7
    const int l    = tid & 31;
    const int ln   = l & 15;
    const int half = (l >> 4) & 1;
    const int t0   = seg * 128;         // time of LDS row 16

    const __bf16* Xb = X + (size_t)b * CIN * S;

    // ---- async weight stage: [co][KV] global -> [co][KP] LDS (16B chunks) --
    {
        const unsigned wbase = lds_addr(&wlds[0]);
        constexpr int CPR = KV / 8;     // 16B chunks per row
        for (int ch = tid; ch < 128 * CPR; ch += 256) {
            const int co = ch / CPR;
            const int cc = (ch % CPR) * 8;
            async_copy_b128(wbase + (unsigned)(co * KP + cc) * 2u,
                            Wt + (size_t)co * KV + cc);
        }
    }

    // ---- stage activations (time-major, causal zero pad at t<0) ----
    for (int idx = tid; idx < CIN * 36; idx += 256) {
        const int c  = idx / 36;
        const int jq = (idx % 36) * 4;      // row quad
        const int tg = t0 - 16 + jq;
        union { uint2 u; __bf16 h[4]; } v;
        if (tg >= 0) {
            v.u = *(const uint2*)(Xb + (size_t)c * S + tg);
        } else {
#pragma unroll
            for (int e = 0; e < 4; ++e) {
                const int te = tg + e;
                v.h[e] = (te >= 0) ? Xb[(size_t)c * S + te] : (__bf16)0.0f;
            }
        }
        __bf16* dst = &act[jq * CPAD + c];
#pragma unroll
        for (int e = 0; e < 4; ++e) dst[e * CPAD] = v.h[e];
    }
    wait_async0();
    __syncthreads();

    // ---- A fragments: 16 time rows for this wave, all K chunks ----
    const int rbase = 16 + w * 16 + ln;     // tap1 LDS row for this lane
    v16bf afrag[NCH];
#pragma unroll
    for (int kc = 0; kc < NCH; ++kc) {
        const int  kb   = kc * 32;
        const bool tap1 = (NTAPS == 1) || (kb >= CIN);
        const int  row  = rbase - (tap1 ? 0 : dil);
        const int  cb   = ((NTAPS == 2 && tap1) ? kb - CIN : kb) + half * 8;
        union { v16bf v; uint4 u[2]; } A;
        const __bf16* ap = &act[row * CPAD + cb];
        A.u[0] = *(const uint4*)ap;          // K {0..7}+halfofs
        A.u[1] = *(const uint4*)(ap + 16);   // K {16..23}+halfofs
        afrag[kc] = A.v;
    }

    const int tg0 = t0 + w * 16 + half * 8;  // first time this lane stores
#pragma unroll
    for (int ct = 0; ct < 8; ++ct) {
        const int co = ct * 16 + ln;
        v8f acc = {};
        const __bf16* wrow = &wlds[co * KP + half * 8];
#pragma unroll
        for (int kc = 0; kc < NCH; ++kc) {
            union { v16bf v; uint4 u[2]; } B;
            const __bf16* wp = wrow + kc * 32;
            B.u[0] = *(const uint4*)wp;
            B.u[1] = *(const uint4*)(wp + 16);
            acc = __builtin_amdgcn_wmma_f32_16x16x32_bf16(
                false, afrag[kc], false, B.v, (short)0, acc, false, false);
        }
        // ---- epilogue: bias / ELU / residual / skip ----
        const float bv = bias[co];
        float r[8];
#pragma unroll
        for (int e = 0; e < 8; ++e) r[e] = acc[e] + bv;
        if (mode >= 1) {
#pragma unroll
            for (int e = 0; e < 8; ++e) r[e] = eluf(r[e]);
        }
        const size_t ybase = ((size_t)b * 128 + co) * S + tg0;
        if (mode >= 2) {
            union { uint4 u; __bf16 h[8]; } rv;
            rv.u = *(const uint4*)(Res + ybase);
#pragma unroll
            for (int e = 0; e < 8; ++e) r[e] = eluf(r[e] + (float)rv.h[e]);
        }
        if (mode == 3) {
            union { uint4 u; __bf16 h[8]; } sv;
            sv.u = *(const uint4*)(Skip + ybase);
#pragma unroll
            for (int e = 0; e < 8; ++e) r[e] += (float)sv.h[e];
        }
        if (Yb) {
            union { uint4 u; __bf16 h[8]; } ov;
#pragma unroll
            for (int e = 0; e < 8; ++e) ov.h[e] = (__bf16)r[e];
            *(uint4*)(Yb + ybase) = ov.u;
        }
        if (Yf) {
            *(float4*)(Yf + ybase)     = make_float4(r[0], r[1], r[2], r[3]);
            *(float4*)(Yf + ybase + 4) = make_float4(r[4], r[5], r[6], r[7]);
        }
    }
}

// ---------------------------------------------------------------------------
extern "C" void kernel_launch(void* const* d_in, const int* in_sizes, int n_in,
                              void* d_out, int out_size, void* d_ws, size_t ws_size,
                              hipStream_t stream) {
    (void)n_in; (void)out_size; (void)ws_size;
    const int B = 16;
    const int S = in_sizes[0] / (B * 64);

    const float* x    = (const float*)d_in[0];
    const float* w1_0 = (const float*)d_in[1];
    const float* b1_0 = (const float*)d_in[2];
    const float* w2_0 = (const float*)d_in[3];
    const float* b2_0 = (const float*)d_in[4];
    const float* ds_w = (const float*)d_in[5];
    const float* ds_b = (const float*)d_in[6];
    const float* W1   = (const float*)d_in[7];
    const float* B1   = (const float*)d_in[8];
    const float* W2   = (const float*)d_in[9];
    const float* B2   = (const float*)d_in[10];

    char* ws = (char*)d_ws;
    size_t off = 0;
    auto alloc = [&](size_t bytes) -> char* {
        char* p = ws + off;
        off += (bytes + 255) & ~(size_t)255;
        return p;
    };
    const size_t actB = (size_t)B * 128 * S * sizeof(__bf16);
    __bf16* xb0 = (__bf16*)alloc((size_t)B * 64 * S * sizeof(__bf16));
    __bf16* Tb  = (__bf16*)alloc(actB);
    __bf16* Rb  = (__bf16*)alloc(actB);
    __bf16* X1b = (__bf16*)alloc(actB);   // block-0 output, kept for final skip
    __bf16* PA  = (__bf16*)alloc(actB);
    __bf16* PB  = (__bf16*)alloc(actB);
    __bf16* wds = (__bf16*)alloc(128 * 64  * sizeof(__bf16));
    __bf16* w1p = (__bf16*)alloc(128 * 128 * sizeof(__bf16));
    __bf16* w2p = (__bf16*)alloc(128 * 256 * sizeof(__bf16));
    __bf16* Wp[8];
    for (int i = 0; i < 8; ++i) Wp[i] = (__bf16*)alloc(128 * 256 * sizeof(__bf16));

    // ---- prep: bf16 input copy + weight repacks ----
    f32_to_bf16_kernel<<<2048, 256, 0, stream>>>(x, xb0, B * 64 * S);
    prep_w_kernel<<<64,  256, 0, stream>>>(ds_w, wds, 64, 1);
    prep_w_kernel<<<64,  256, 0, stream>>>(w1_0, w1p, 64, 2);
    prep_w_kernel<<<128, 256, 0, stream>>>(w2_0, w2p, 128, 2);
    for (int i = 0; i < 4; ++i) {
        prep_w_kernel<<<128, 256, 0, stream>>>(W1 + (size_t)i * 128 * 128 * 2, Wp[i],     128, 2);
        prep_w_kernel<<<128, 256, 0, stream>>>(W2 + (size_t)i * 128 * 128 * 2, Wp[4 + i], 128, 2);
    }

    dim3 grid(S / 128, B), blk(256);   // 32 x 16 = 512 workgroups
    // ---- block 0 (64 -> 128, dilation 1, 1x1 downsample residual) ----
    tcn_conv_kernel<64, 1><<<grid, blk, 0, stream>>>(
        xb0, wds, ds_b, nullptr, nullptr, Rb, nullptr, S, 1, 0);        // ds conv
    tcn_conv_kernel<64, 2><<<grid, blk, 0, stream>>>(
        xb0, w1p, b1_0, nullptr, nullptr, Tb, nullptr, S, 1, 1);        // conv1+ELU
    tcn_conv_kernel<128, 2><<<grid, blk, 0, stream>>>(
        Tb, w2p, b2_0, Rb, nullptr, X1b, nullptr, S, 1, 2);             // conv2+res

    // ---- blocks 1..4 (128 -> 128, dilation 2^i) ----
    const __bf16* cur = X1b;
    __bf16* outs[3] = {PA, PB, PA};
    for (int i = 1; i <= 4; ++i) {
        const int d = 1 << i;
        tcn_conv_kernel<128, 2><<<grid, blk, 0, stream>>>(
            cur, Wp[i - 1], B1 + (size_t)(i - 1) * 128,
            nullptr, nullptr, Tb, nullptr, S, d, 1);                    // conv1+ELU
        if (i < 4) {
            __bf16* o = outs[i - 1];
            tcn_conv_kernel<128, 2><<<grid, blk, 0, stream>>>(
                Tb, Wp[4 + i - 1], B2 + (size_t)(i - 1) * 128,
                cur, nullptr, o, nullptr, S, d, 2);                     // conv2+res
            cur = o;
        } else {
            // final layer: conv2 + residual + skip(outputs[1]) -> fp32 d_out
            tcn_conv_kernel<128, 2><<<grid, blk, 0, stream>>>(
                Tb, Wp[7], B2 + 3 * 128,
                cur, X1b, nullptr, (float*)d_out, S, d, 3);
        }
    }
}
